// SegmentErrorRateLoss_87299505258881
// MI455X (gfx1250) — 8-bit matrix operand order repaired and checked by reading, compiled, not run on hardware
//
#include <hip/hip_runtime.h>
#include <hip/hip_bf16.h>

typedef __attribute__((ext_vector_type(8))) int v8i;
union V8 { v8i v; int e[8]; };

#define BB   128
#define CC   128
#define TT   2048
#define NTH  256
#define CHK  9

__global__ __launch_bounds__(256) void argmax_kernel(const float* __restrict__ in,
                                                     unsigned short* __restrict__ pred) {
    int idx4 = blockIdx.x * 256 + threadIdx.x;
    int b    = idx4 >> 9;
    int tq   = idx4 & 511;
    const float4* base = (const float4*)(in + (size_t)b * (CC * TT));

    float4 v = base[tq];
    float bv0 = v.x, bv1 = v.y, bv2 = v.z, bv3 = v.w;
    int   bi0 = 0,   bi1 = 0,   bi2 = 0,   bi3 = 0;
    for (int c = 1; c < CC; ++c) {
        float4 u = base[c * (TT / 4) + tq];
        if (u.x > bv0) { bv0 = u.x; bi0 = c; }
        if (u.y > bv1) { bv1 = u.y; bi1 = c; }
        if (u.z > bv2) { bv2 = u.z; bi2 = c; }
        if (u.w > bv3) { bv3 = u.w; bi3 = c; }
    }
    unsigned long long pk = (unsigned long long)(unsigned short)bi0
                          | ((unsigned long long)(unsigned short)bi1 << 16)
                          | ((unsigned long long)(unsigned short)bi2 << 32)
                          | ((unsigned long long)(unsigned short)bi3 << 48);
    ((unsigned long long*)pred)[idx4] = pk;
}

__global__ __launch_bounds__(256) void collapse_kernel(const unsigned short* __restrict__ pred,
                                                       const int* __restrict__ target,
                                                       unsigned short* __restrict__ a_seq,
                                                       unsigned short* __restrict__ b_seq,
                                                       int* __restrict__ m_len,
                                                       int* __restrict__ n_len) {
    __shared__ int s_wsum[8];
    const int b = blockIdx.x;
    const int tid = threadIdx.x;
    const int lane = tid & 31, wv = tid >> 5;

    for (int pass = 0; pass < 2; ++pass) {
        int e0 = tid * 8;
        int vals[8];
        int prev;
        if (pass == 0) {
            #pragma unroll
            for (int k = 0; k < 8; ++k) vals[k] = (int)pred[b * TT + e0 + k];
            prev = (e0 > 0) ? (int)pred[b * TT + e0 - 1] : -1;
        } else {
            #pragma unroll
            for (int k = 0; k < 8; ++k) vals[k] = target[b * TT + e0 + k];
            prev = (e0 > 0) ? target[b * TT + e0 - 1] : -1;
        }
        int kept[8]; int cnt = 0; int p = prev;
        #pragma unroll
        for (int k = 0; k < 8; ++k) {
            int x = vals[k];
            bool changed = ((e0 + k) == 0) ? true : (x != p);
            if (changed && (x != 0)) kept[cnt++] = x;
            p = x;
        }
        int inc = cnt;
        for (int off = 1; off < 32; off <<= 1) {
            int y = __shfl_up(inc, off, 32);
            if (lane >= off) inc += y;
        }
        if (lane == 31) s_wsum[wv] = inc;
        __syncthreads();
        int woff = 0, total = 0;
        for (int w = 0; w < 8; ++w) { if (w < wv) woff += s_wsum[w]; total += s_wsum[w]; }
        int base = woff + (inc - cnt);
        unsigned short* dst = (pass == 0) ? a_seq : b_seq;
        for (int k = 0; k < cnt; ++k) dst[b * TT + base + k] = (unsigned short)kept[k];
        if (tid == 0) { if (pass == 0) m_len[b] = total; else n_len[b] = total; }
        __syncthreads();
    }
}

__global__ __launch_bounds__(256) void dp_kernel(const unsigned short* __restrict__ a_seq,
                                                 const unsigned short* __restrict__ b_seq,
                                                 const int* __restrict__ m_len,
                                                 const int* __restrict__ n_len,
                                                 float* __restrict__ losses) {
    __shared__ int s_prev[TT + 1];
    __shared__ int s_cur[TT + 1];
    __shared__ int s_eq32[TT * 4];
    __shared__ int s_aval[16];
    __shared__ unsigned short s_bval[TT];
    __shared__ int s_wmin[8];
    __shared__ int s_mn[2];

    unsigned char* s_eq = (unsigned char*)s_eq32;

    const int bidx = blockIdx.x;
    const int tid  = threadIdx.x;
    const int lane = tid & 31;
    const int wv   = tid >> 5;
    const int INF  = 0x3FFFFFFF;

    if (tid == 0) { s_mn[0] = m_len[bidx]; s_mn[1] = n_len[bidx]; }
    for (int j = tid; j < TT; j += NTH) s_bval[j] = b_seq[bidx * TT + j];
    __syncthreads();
    const int m = s_mn[0], n = s_mn[1];

    for (int j = tid; j <= n; j += NTH) s_prev[j] = j;

    int* prevp = s_prev;
    int* curp  = s_cur;

    const int ncols = n + 1;
    const int chunk = (ncols + NTH - 1) / NTH;
    const int jbeg  = tid * chunk;
    const int jend  = (jbeg + chunk < ncols) ? (jbeg + chunk) : ncols;
    const int ntile = (n + 15) >> 4;

    for (int i0 = 0; i0 < m; i0 += 16) {
        if (tid < 16) s_aval[tid] = (int)a_seq[bidx * TT + i0 + tid];
        __syncthreads();

        {
            const int row = lane & 15;
            const int hi  = (lane >> 4) & 1;
            const int av  = s_aval[row];
            V8 A, Cz;
            #pragma unroll
            for (int q = 0; q < 8; ++q) {
                int kl  = (q >> 1) * 16 + hi * 8 + (q & 1) * 4;
                int rel = av - kl;
                A.e[q]  = (rel >= 0 && rel < 4) ? (1 << (rel * 8)) : 0;
                Cz.e[q] = 0;
            }
            for (int tile = wv; tile < ntile; tile += 8) {
                int j0  = tile << 4;
                int col = lane & 15;
                int bvv = (int)s_bval[j0 + col];
                V8 Bm;
                #pragma unroll
                for (int q = 0; q < 8; ++q) {
                    int kb  = (q >> 1) * 16 + hi * 8 + (q & 1) * 4;
                    int rel = bvv - kb;
                    Bm.e[q] = (rel >= 0 && rel < 4) ? (1 << (rel * 8)) : 0;
                }
                V8 D;
                D.v = __builtin_amdgcn_wmma_i32_16x16x64_iu8(false, A.v, false, Bm.v,
                                                             Cz.v, false, false);
                int N  = lane & 15;
                int Mb = hi ? 8 : 0;
                int pk0 = D.e[0] | (D.e[1] << 8) | (D.e[2] << 16) | (D.e[3] << 24);
                int pk1 = D.e[4] | (D.e[5] << 8) | (D.e[6] << 16) | (D.e[7] << 24);
                int ca  = (j0 + N) * 16 + Mb;
                *(int*)(s_eq + ca)     = pk0;
                *(int*)(s_eq + ca + 4) = pk1;
            }
        }
        __syncthreads();

        const int rmax = (m - i0 < 16) ? (m - i0) : 16;
        for (int r = 0; r < rmax; ++r) {
            int loc[CHK];
            int run = INF;
            for (int k = 0; k < chunk; ++k) {
                int j = jbeg + k;
                if (j < jend) {
                    int d;
                    if (j == 0) d = prevp[0] + 1;
                    else {
                        int eq  = (int)s_eq[(j - 1) * 16 + r];
                        int del = prevp[j];
                        int sub = prevp[j - 1] - eq;
                        d = ((del < sub) ? del : sub) + 1;
                    }
                    d -= j;
                    run = (d < run) ? d : run;
                    loc[k] = run;
                }
            }
            int inc = run;
            for (int off = 1; off < 32; off <<= 1) {
                int y = __shfl_up(inc, off, 32);
                if (lane >= off) inc = (y < inc) ? y : inc;
            }
            if (lane == 31) s_wmin[wv] = inc;
            int excl = __shfl_up(inc, 1, 32);
            if (lane == 0) excl = INF;
            __syncthreads();
            int pre = excl;
            for (int w = 0; w < 8; ++w)
                if (w < wv && s_wmin[w] < pre) pre = s_wmin[w];
            for (int k = 0; k < chunk; ++k) {
                int j = jbeg + k;
                if (j < jend) {
                    int s = (loc[k] < pre) ? loc[k] : pre;
                    curp[j] = s + j;
                }
            }
            __syncthreads();
            int* t = prevp; prevp = curp; curp = t;
        }
    }

    __syncthreads();
    if (tid == 0) {
        float dist  = (float)prevp[n];
        float denom = (float)((n > 1) ? n : 1);
        losses[bidx] = dist / denom;
    }
}

__global__ void reduce_kernel(const float* __restrict__ losses, float* __restrict__ out) {
    __shared__ float s[BB];
    int tid = threadIdx.x;
    s[tid] = losses[tid];
    __syncthreads();
    for (int off = BB / 2; off > 0; off >>= 1) {
        if (tid < off) s[tid] += s[tid + off];
        __syncthreads();
    }
    if (tid == 0) out[0] = s[0] * (1.0f / (float)BB);
}

extern "C" void kernel_launch(void* const* d_in, const int* in_sizes, int n_in,
                              void* d_out, int out_size, void* d_ws, size_t ws_size,
                              hipStream_t stream) {
    (void)in_sizes; (void)n_in; (void)out_size; (void)ws_size;
    const float* inp    = (const float*)d_in[0];
    const int*   target = (const int*)d_in[1];
    float*       out    = (float*)d_out;

    char* ws = (char*)d_ws;
    unsigned short* pred   = (unsigned short*)(ws);
    unsigned short* a_seq  = (unsigned short*)(ws + 524288);
    unsigned short* b_seq  = (unsigned short*)(ws + 1048576);
    int*            m_len  = (int*)(ws + 1572864);
    int*            n_len  = (int*)(ws + 1573376);
    float*          losses = (float*)(ws + 1573888);

    argmax_kernel<<<(BB * TT / 4) / 256, 256, 0, stream>>>(inp, pred);
    collapse_kernel<<<BB, NTH, 0, stream>>>(pred, target, a_seq, b_seq, m_len, n_len);
    dp_kernel<<<BB, NTH, 0, stream>>>(a_seq, b_seq, m_len, n_len, losses);
    reduce_kernel<<<1, BB, 0, stream>>>(losses, out);
}
